// Memory_11227044511945
// MI455X (gfx1250) — matrix operand order repaired and checked by reading, not compiled, not run on hardware
//
#include <hip/hip_runtime.h>
#include <hip/hip_bf16.h>
#include <stdint.h>

#define N_Q   32768
#define M_IT  2048
#define D_DIM 512
#define QPB   32      // query rows per workgroup (two WMMA M-tiles)
#define STR   2056    // LDS score row stride (bf16): 16B-aligned rows

typedef __attribute__((ext_vector_type(16))) __bf16       v16bf;
typedef __attribute__((ext_vector_type(8)))  float        v8f;
typedef __attribute__((ext_vector_type(4)))  float        f4;
typedef __attribute__((ext_vector_type(4)))  unsigned int u4;
typedef __attribute__((ext_vector_type(4)))  __bf16       bf4;

union ABFrag { v16bf v; u4 u[2]; };

// ---------------- fp32 -> bf16 conversion --------------------------------
__global__ void cvt_bf16(const float* __restrict__ src, __bf16* __restrict__ dst, int n) {
  int i = (blockIdx.x * blockDim.x + threadIdx.x) * 4;
  if (i < n) {
    f4 v = *(const f4*)(src + i);
    bf4 r;
    r.x = (__bf16)v.x; r.y = (__bf16)v.y; r.z = (__bf16)v.z; r.w = (__bf16)v.w;
    *(bf4*)(dst + i) = r;
  }
}

// ---------------- fused: bf16 WMMA GEMM + softmax + top10 + gather -------
__global__ __launch_bounds__(256)
void memory_fused(const __bf16* __restrict__ qbf, const __bf16* __restrict__ mbf,
                  const float* __restrict__ query, const float* __restrict__ mitems,
                  float* __restrict__ out) {
  __shared__ __bf16 S16[QPB * STR];   // 32 rows x 2048 bf16 scores (padded)

  const int tid   = threadIdx.x;
  const int lane  = tid & 31;
  const int w     = tid >> 5;        // wave 0..7
  const int mgrp  = w >> 2;          // 0/1: which 16-query M-tile
  const int ngrp  = w & 3;           // 0..3: which 512-column slice
  const int hlf   = lane >> 4;
  const int m16   = lane & 15;
  const int qbase = blockIdx.x * QPB;

  // ===== Phase 1: scores[32 x 2048] = Q_bf16 x M_bf16^T (f32 accum) =====
  v16bf afrag[16];
  {
    const __bf16* qrow = qbf + (size_t)(qbase + mgrp * 16 + m16) * D_DIM + hlf * 16;
    #pragma unroll
    for (int kt = 0; kt < 16; ++kt) {
      ABFrag a;
      a.u[0] = *(const u4*)(qrow + kt * 32);
      a.u[1] = *(const u4*)(qrow + kt * 32 + 8);
      afrag[kt] = a.v;
    }
  }

  const int wcol = ngrp * 512;       // this wave's 512 score columns
  const __bf16* b0 = mbf + (size_t)(wcol + m16) * D_DIM + hlf * 16;
  const __bf16* b1 = mbf + (size_t)(wcol + 16 + m16) * D_DIM + hlf * 16;

  // depth-4 B ring, continuous across all 16 column-tile pairs
  ABFrag ba[4], bb[4];
  #pragma unroll
  for (int k = 0; k < 4; ++k) {
    ba[k].u[0] = *(const u4*)(b0 + k * 32); ba[k].u[1] = *(const u4*)(b0 + k * 32 + 8);
    bb[k].u[0] = *(const u4*)(b1 + k * 32); bb[k].u[1] = *(const u4*)(b1 + k * 32 + 8);
  }

  #pragma unroll 1
  for (int ntp = 0; ntp < 16; ++ntp) {
    const int n0 = wcol + ntp * 32;
    v8f c0 = {0.f,0.f,0.f,0.f,0.f,0.f,0.f,0.f};
    v8f c1 = {0.f,0.f,0.f,0.f,0.f,0.f,0.f,0.f};

    #pragma unroll
    for (int kt = 0; kt < 16; ++kt) {
      const int cur = kt & 3;
      c0 = __builtin_amdgcn_wmma_f32_16x16x32_bf16(false, afrag[kt], false, ba[cur].v,
                                                   (short)0, c0, false, false);
      c1 = __builtin_amdgcn_wmma_f32_16x16x32_bf16(false, afrag[kt], false, bb[cur].v,
                                                   (short)0, c1, false, false);
      __builtin_amdgcn_sched_group_barrier(0x008, 2, 0);   // pin: 2 WMMA
      if (kt < 12) {                                       // refill ring: this pair
        ba[cur].u[0] = *(const u4*)(b0 + (kt + 4) * 32);
        ba[cur].u[1] = *(const u4*)(b0 + (kt + 4) * 32 + 8);
        bb[cur].u[0] = *(const u4*)(b1 + (kt + 4) * 32);
        bb[cur].u[1] = *(const u4*)(b1 + (kt + 4) * 32 + 8);
        __builtin_amdgcn_sched_group_barrier(0x020, 4, 0); // then: 4 VMEM loads
      } else if (ntp + 1 < 16) {                           // refill: next pair kt 0..3
        const __bf16* nb0 = b0 + 32 * D_DIM + (kt - 12) * 32;
        const __bf16* nb1 = b1 + 32 * D_DIM + (kt - 12) * 32;
        ba[cur].u[0] = *(const u4*)(nb0);
        ba[cur].u[1] = *(const u4*)(nb0 + 8);
        bb[cur].u[0] = *(const u4*)(nb1);
        bb[cur].u[1] = *(const u4*)(nb1 + 8);
        __builtin_amdgcn_sched_group_barrier(0x020, 4, 0);
      }
    }

    // C layout: VGPR v -> row v (lanes 0-15) / row v+8 (lanes 16-31)
    #pragma unroll
    for (int v = 0; v < 8; ++v) {
      const int r = mgrp * 16 + v + hlf * 8;
      S16[r * STR + n0 + m16]      = (__bf16)c0[v];
      S16[r * STR + n0 + 16 + m16] = (__bf16)c1[v];
    }
    b0 += 32 * D_DIM;
    b1 += 32 * D_DIM;
  }
  __syncthreads();

  // ===== Phase 2+3: per wave 4 rows: stats, top-10 (int-key argmax), gather ==
  #pragma unroll 1
  for (int rr = 0; rr < 4; ++rr) {
    const int row  = w * 4 + rr;
    const int grow = qbase + row;
    const __bf16* Srow = &S16[row * STR];

    // lane owns idx = t*256 + lane*8 + {0..7}; 8x 16B packed loads
    u4 pk[8];
    #pragma unroll
    for (int t = 0; t < 8; ++t)
      pk[t] = *(const u4*)(Srow + t * 256 + lane * 8);

    // order-preserving keys: (ordered bf16 <<11) | (2047 - idx); unique keys,
    // umax == argmax with lowest-index tie-break (like lax.top_k)
    unsigned int key[64];
    #pragma unroll
    for (int t = 0; t < 8; ++t) {
      #pragma unroll
      for (int q = 0; q < 4; ++q) {
        const unsigned int u = pk[t][q];
        const int i0 = t * 256 + lane * 8 + q * 2;
        unsigned int blo = u & 0xffffu, bhi = u >> 16;
        blo = (blo & 0x8000u) ? (0xffffu - blo) : (blo | 0x8000u);
        bhi = (bhi & 0x8000u) ? (0xffffu - bhi) : (bhi | 0x8000u);
        key[t * 8 + q * 2]     = (blo << 11) | (unsigned)(2047 - i0);
        key[t * 8 + q * 2 + 1] = (bhi << 11) | (unsigned)(2046 - i0);
      }
    }

    float p[10]; int pid[10];
    float mx = 0.f, invZ = 0.f;
    #pragma unroll
    for (int j = 0; j < 10; ++j) {
      unsigned int acc[8];
      #pragma unroll
      for (int t = 0; t < 8; ++t) {
        unsigned int a = key[t * 8];
        #pragma unroll
        for (int e = 1; e < 8; ++e) a = a > key[t * 8 + e] ? a : key[t * 8 + e];
        acc[t] = a;
      }
      unsigned int bk = acc[0];
      #pragma unroll
      for (int t = 1; t < 8; ++t) bk = bk > acc[t] ? bk : acc[t];
      #pragma unroll
      for (int off = 16; off; off >>= 1) {
        const unsigned int ok = (unsigned)__shfl_xor((int)bk, off, 32);
        bk = bk > ok ? bk : ok;
      }
      // decode winner
      const int idx = 2047 - (int)(bk & 0x7ffu);
      const unsigned int ob = bk >> 11;
      const unsigned int fbits =
          ((ob & 0x8000u) ? (ob & 0x7fffu) : (0xffffu - ob)) << 16;
      const float v = __int_as_float((int)fbits);
      if (j == 0) {            // top-1 value == row max; compute Z once
        mx = v;
        float Z = 0.f;
        #pragma unroll
        for (int t = 0; t < 8; ++t) {
          #pragma unroll
          for (int q = 0; q < 4; ++q) {
            const unsigned int u = pk[t][q];
            Z += __expf(__int_as_float((int)(u << 16)) - mx);
            Z += __expf(__int_as_float((int)(u & 0xffff0000u)) - mx);
          }
        }
        #pragma unroll
        for (int off = 16; off; off >>= 1) Z += __shfl_xor(Z, off, 32);
        invZ = 1.f / Z;
      }
      p[j]  = __expf(v - mx) * invZ;   // softmax probability
      pid[j] = idx;
      #pragma unroll
      for (int e = 0; e < 64; ++e) key[e] = (key[e] == bk) ? 0u : key[e];
    }

    // second-stage softmax over probabilities (p sorted descending)
    float et[5], eb[5], st = 0.f, sb = 0.f;
    #pragma unroll
    for (int k = 0; k < 5; ++k) { et[k] = __expf(p[k]     - p[0]); st += et[k]; }
    #pragma unroll
    for (int k = 0; k < 5; ++k) { eb[k] = __expf(p[5 + k] - p[5]); sb += eb[k]; }
    const float ist = 1.f / st, isb = 1.f / sb;

    // gather + blend in fp32: each lane handles 16 of 512 dims
    const int d0 = lane * 16;
    const float* qp = query + (size_t)grow * D_DIM + d0;
    f4 qa0 = *(const f4*)(qp + 0),  qa1 = *(const f4*)(qp + 4);
    f4 qa2 = *(const f4*)(qp + 8),  qa3 = *(const f4*)(qp + 12);
    f4 at0 = {0,0,0,0}, at1 = {0,0,0,0}, at2 = {0,0,0,0}, at3 = {0,0,0,0};
    f4 ab0 = {0,0,0,0}, ab1 = {0,0,0,0}, ab2 = {0,0,0,0}, ab3 = {0,0,0,0};
    #pragma unroll
    for (int k = 0; k < 5; ++k) {
      const float* kr = mitems + (size_t)pid[k] * D_DIM + d0;
      const float wk = et[k] * ist;
      at0 += wk * *(const f4*)(kr + 0);
      at1 += wk * *(const f4*)(kr + 4);
      at2 += wk * *(const f4*)(kr + 8);
      at3 += wk * *(const f4*)(kr + 12);
    }
    #pragma unroll
    for (int k = 0; k < 5; ++k) {
      const float* kr = mitems + (size_t)pid[5 + k] * D_DIM + d0;
      const float wk = eb[k] * isb;
      ab0 += wk * *(const f4*)(kr + 0);
      ab1 += wk * *(const f4*)(kr + 4);
      ab2 += wk * *(const f4*)(kr + 8);
      ab3 += wk * *(const f4*)(kr + 12);
    }
    float* on = out + (size_t)grow * D_DIM + d0;
    float* oa = out + (size_t)N_Q * D_DIM + (size_t)grow * D_DIM + d0;
    *(f4*)(on + 0)  = 0.5f  * qa0 + 0.5f  * at0;
    *(f4*)(on + 4)  = 0.5f  * qa1 + 0.5f  * at1;
    *(f4*)(on + 8)  = 0.5f  * qa2 + 0.5f  * at2;
    *(f4*)(on + 12) = 0.5f  * qa3 + 0.5f  * at3;
    *(f4*)(oa + 0)  = 0.01f * qa0 + 0.99f * ab0;
    *(f4*)(oa + 4)  = 0.01f * qa1 + 0.99f * ab1;
    *(f4*)(oa + 8)  = 0.01f * qa2 + 0.99f * ab2;
    *(f4*)(oa + 12) = 0.01f * qa3 + 0.99f * ab3;
  }
}

extern "C" void kernel_launch(void* const* d_in, const int* in_sizes, int n_in,
                              void* d_out, int out_size, void* d_ws, size_t ws_size,
                              hipStream_t stream) {
  const float* query  = (const float*)d_in[0];   // (32768, 512) fp32
  const float* mitems = (const float*)d_in[1];   // (2048, 512) fp32 (unit rows)
  float* out = (float*)d_out;                    // pseudo_normal then pseudo_abnormal

  __bf16* qbf = (__bf16*)d_ws;                         // 32 MB
  __bf16* mbf = qbf + (size_t)N_Q * D_DIM;             // +2 MB

  const int nq = N_Q * D_DIM, nm = M_IT * D_DIM;
  cvt_bf16<<<(nq / 4 + 255) / 256, 256, 0, stream>>>(query,  qbf, nq);
  cvt_bf16<<<(nm / 4 + 255) / 256, 256, 0, stream>>>(mitems, mbf, nm);
  memory_fused<<<N_Q / QPB, 256, 0, stream>>>(qbf, mbf, query, mitems, out);
}
